// _GEM_37623913513352
// MI455X (gfx1250) — matrix operand order repaired and checked by reading, not compiled, not run on hardware
//
#include <hip/hip_runtime.h>
#include <hip/hip_bf16.h>

typedef __bf16 bf16;
typedef __attribute__((ext_vector_type(8)))  __bf16 v8bf;
typedef __attribute__((ext_vector_type(16))) __bf16 v16bf;
typedef __attribute__((ext_vector_type(8)))  float  v8f;
typedef int v4i __attribute__((ext_vector_type(4)));
typedef __attribute__((address_space(1))) void as1_void;
typedef __attribute__((address_space(3))) void as3_void;
typedef __attribute__((address_space(1))) v4i  as1_v4i;
typedef __attribute__((address_space(3))) v4i  as3_v4i;

// CDNA5 async global->LDS copy (ASYNCcnt), guarded so both toolchains compile.
#if defined(__AMDGCN__) && __has_builtin(__builtin_amdgcn_global_load_async_to_lds_b128)
#define ASYNC_LDS 1
#else
#define ASYNC_LDS 0
#endif

// Problem constants
constexpr int kB = 16, kN = 12, kD = 49, kC = 512, kH = 256, kP = 144; // kP = N*N
constexpr float kScale = 0.0625f;  // H^-0.5 = 1/16
constexpr float kEps = 1e-5f;
constexpr int kSBS = 520;   // GEMM B-tile LDS row stride (bank-conflict-free)
constexpr int kSS  = 66;    // score matrix LDS row stride (f32)
constexpr int kPS  = 72;    // prob matrix LDS row stride (bf16)
constexpr int kKP  = 64;    // padded attention K (49 -> 64)

__device__ __forceinline__ v16bf make_frag(v8bf lo, v8bf hi) {
  return __builtin_shufflevector(lo, hi,
      0, 1, 2, 3, 4, 5, 6, 7, 8, 9, 10, 11, 12, 13, 14, 15);
}
// A fragment (16-bit, 16x32): per-lane two contiguous 8-elem groups.
__device__ __forceinline__ v16bf load_afrag(const bf16* row, int kb, int half) {
  return make_frag(*(const v8bf*)(row + kb + half * 8),
                   *(const v8bf*)(row + kb + 16 + half * 8));
}
// B fragment (16-bit, 32x16): per-lane two contiguous 8-elem K groups of its column.
__device__ __forceinline__ v16bf load_bfrag(const bf16* row, int kb, int half) {
  return make_frag(*(const v8bf*)(row + kb + half * 8),
                   *(const v8bf*)(row + kb + 16 + half * 8));
}
__device__ __forceinline__ v8f wmma_bf16(v16bf a, v16bf b, v8f c) {
  return __builtin_amdgcn_wmma_f32_16x16x32_bf16(false, a, false, b,
                                                 (short)0, c, false, false);
}

// ---------------------------------------------------------------------------
// f32 -> bf16 conversion
// ---------------------------------------------------------------------------
__global__ __launch_bounds__(256) void cvt_bf16(const float* __restrict__ src,
                                                bf16* __restrict__ dst, int n) {
  int i = blockIdx.x * 256 + threadIdx.x;
  if (i < n) dst[i] = (bf16)src[i];
}

// ---------------------------------------------------------------------------
// WMMA GEMM: Y(M,Nn) = X(M,512) * W(Nn,512)^T + bias(Nn); optional bf16 copy.
// Block = 8 waves computing a 128(M) x 64(N) tile. B tile staged in LDS
// (async copy when available). K pipeline: fully unrolled ping-pong (no
// loop-carried registers -> no rotation copies; scheduler pipelines freely).
// ---------------------------------------------------------------------------
__global__ __launch_bounds__(256)
void gemm_bf16(const bf16* __restrict__ X, const bf16* __restrict__ W,
               const float* __restrict__ bias, float* __restrict__ Y,
               bf16* __restrict__ Ybf, int M, int Nn) {
  constexpr int K = kC;                          // 512, compile-time
  __shared__ __align__(16) bf16 sB[64 * kSBS];   // 64 cols x 512 K, padded

  const int wave = threadIdx.x >> 5;
  const int lane = threadIdx.x & 31;
  const int half = lane >> 4;
  const int l16  = lane & 15;
  const int col0 = blockIdx.x * 64;
  const int mtiles = M >> 4;
  int tm = blockIdx.y * 8 + wave;
  if (tm >= mtiles) tm = mtiles - 1;    // clamp: duplicates recompute same data

  // ---- stage B tile (64 rows of W, each K bf16) into LDS ----
  {
    constexpr int chunksPerRow = K >> 3;
    constexpr int totalChunks = 64 * chunksPerRow;
    for (int ch = (int)threadIdx.x; ch < totalChunks; ch += 256) {
      const int r = ch / chunksPerRow;
      const int c = (ch % chunksPerRow) << 3;
      bf16* gsrc = const_cast<bf16*>(W) + (size_t)(col0 + r) * K + c;
      bf16* ldst = &sB[r * kSBS + c];
#if ASYNC_LDS
      __builtin_amdgcn_global_load_async_to_lds_b128(
          (as1_v4i*)(as1_void*)gsrc, (as3_v4i*)(as3_void*)ldst, 0, 0);
#else
      *(v8bf*)ldst = *(const v8bf*)gsrc;
#endif
    }
#if ASYNC_LDS
#if __has_builtin(__builtin_amdgcn_s_wait_asynccnt)
    __builtin_amdgcn_s_wait_asynccnt(0);
#else
    asm volatile("s_wait_asynccnt 0x0" ::: "memory");
#endif
#endif
  }
  __syncthreads();

  const bf16* xrow = X + (size_t)(tm * 16 + l16) * K;
  const bf16* srow[4];
#pragma unroll
  for (int t = 0; t < 4; ++t) srow[t] = &sB[(l16 + t * 16) * kSBS];

  v8f acc[4] = {};
  v16bf a0, a1, b0[4], b1[4];

  a0 = load_afrag(xrow, 0, half);
#pragma unroll
  for (int t = 0; t < 4; ++t) b0[t] = load_bfrag(srow[t], 0, half);

#pragma unroll
  for (int kb = 0; kb < K - 64; kb += 64) {
    // prefetch odd step, consume even step
    a1 = load_afrag(xrow, kb + 32, half);
#pragma unroll
    for (int t = 0; t < 4; ++t) b1[t] = load_bfrag(srow[t], kb + 32, half);
#pragma unroll
    for (int t = 0; t < 4; ++t) acc[t] = wmma_bf16(a0, b0[t], acc[t]);
    // prefetch next even step, consume odd step
    a0 = load_afrag(xrow, kb + 64, half);
#pragma unroll
    for (int t = 0; t < 4; ++t) b0[t] = load_bfrag(srow[t], kb + 64, half);
#pragma unroll
    for (int t = 0; t < 4; ++t) acc[t] = wmma_bf16(a1, b1[t], acc[t]);
  }
  // peeled tail (kb = K-64): no further prefetch
  a1 = load_afrag(xrow, K - 32, half);
#pragma unroll
  for (int t = 0; t < 4; ++t) b1[t] = load_bfrag(srow[t], K - 32, half);
#pragma unroll
  for (int t = 0; t < 4; ++t) acc[t] = wmma_bf16(a0, b0[t], acc[t]);
#pragma unroll
  for (int t = 0; t < 4; ++t) acc[t] = wmma_bf16(a1, b1[t], acc[t]);

#pragma unroll
  for (int t = 0; t < 4; ++t) {
    const int col = col0 + l16 + t * 16;
    const float bv = bias[col];
#pragma unroll
    for (int r = 0; r < 8; ++r) {
      const int row = tm * 16 + half * 8 + r;
      const float val = acc[t][r] + bv;
      if (Y)   Y[(size_t)row * Nn + col] = val;
      if (Ybf) Ybf[(size_t)row * Nn + col] = (bf16)val;
    }
  }
}

// ---------------------------------------------------------------------------
// Wave (32-lane) sum reduction
// ---------------------------------------------------------------------------
__device__ __forceinline__ float wave_sum(float v) {
#pragma unroll
  for (int off = 16; off > 0; off >>= 1) v += __shfl_xor(v, off, 32);
  return v;
}

// ---------------------------------------------------------------------------
// FAM attention: one block per (b, n, dq) query row (small: stays VALU).
// ---------------------------------------------------------------------------
__global__ __launch_bounds__(256)
void fam_attn(const float* __restrict__ q1, const float* __restrict__ k1,
              const float* __restrict__ v1, bf16* __restrict__ feat) {
  const int bq = blockIdx.x;                 // flattened (b, n, dq)
  const int b = bq / (kN * kD);
  const float* qrow = q1 + (size_t)bq * kH;
  const float* kbase = k1 + (size_t)b * kD * kH;
  const float* vbase = v1 + (size_t)b * kD * kC;

  __shared__ float sc[kD];
  const int wave = threadIdx.x >> 5, lane = threadIdx.x & 31;

  for (int j = wave; j < kD; j += 8) {
    const float* krow = kbase + (size_t)j * kH;
    float s = 0.f;
    for (int h = lane; h < kH; h += 32) s += qrow[h] * krow[h];
    s = wave_sum(s);
    if (lane == 0) sc[j] = s * kScale;
  }
  __syncthreads();

  float mx = -1e30f;
  for (int j = 0; j < kD; ++j) mx = fmaxf(mx, sc[j]);
  __syncthreads();
  if (threadIdx.x < kD) sc[threadIdx.x] = expf(sc[threadIdx.x] - mx);
  __syncthreads();
  float ssum = 0.f;
  for (int j = 0; j < kD; ++j) ssum += sc[j];
  const float inv = 1.f / ssum;

  for (int e = threadIdx.x; e < kC; e += 256) {
    float acc = 0.f;
    for (int j = 0; j < kD; ++j) acc += sc[j] * vbase[(size_t)j * kC + e];
    feat[(size_t)bq * kC + e] = (bf16)(acc * inv);
  }
}

// ---------------------------------------------------------------------------
// Transpose + pad V for the ARM AV product: v2 (b,i)(49 x 512) f32 ->
// v2t (b,i)(512 x 64) bf16, K rows 49..63 zero.
// ---------------------------------------------------------------------------
__global__ __launch_bounds__(256)
void transpose_v2(const float* __restrict__ v2, bf16* __restrict__ v2t) {
  const float* src = v2 + (size_t)blockIdx.x * kD * kC;
  bf16* dst = v2t + (size_t)blockIdx.x * kC * kKP;
  for (int idx = threadIdx.x; idx < kC * kKP; idx += 256) {
    const int e = idx >> 6, k = idx & (kKP - 1);
    dst[idx] = (k < kD) ? (bf16)src[(size_t)k * kC + e] : (bf16)0.f;
  }
}

// ---------------------------------------------------------------------------
// ARM pairwise attention, fully WMMA. One block per (b,i,j).
// Stage 1: S = Q K^T (49x49 padded to 64x64) -> LDS (f32).
// Stage 2: row softmax; probs -> LDS bf16 with padded columns zeroed.
// Stage 3: Out = P V via WMMA using pre-transposed v2t; store rows < 49.
// ---------------------------------------------------------------------------
__global__ __launch_bounds__(256)
void arm_attn_wmma(const bf16* __restrict__ q2b, const bf16* __restrict__ k2b,
                   const bf16* __restrict__ v2t, bf16* __restrict__ feat2) {
  __shared__ __align__(16) float S[64 * kSS];     // ~16.5 KB
  __shared__ __align__(16) bf16 Pb[64 * kPS];     //  ~9 KB

  const int bij = blockIdx.x;                // b*144 + i*12 + j
  const int b = bij / kP;
  const int rem = bij % kP;
  const int i = rem / kN;
  const int j = rem % kN;
  const bf16* qs = q2b + (size_t)(b * kN + j) * kD * kH;
  const bf16* ks = k2b + (size_t)(b * kN + i) * kD * kH;
  const bf16* vt = v2t + (size_t)(b * kN + i) * kC * kKP;
  bf16* os = feat2 + (size_t)bij * kD * kC;

  const int wave = threadIdx.x >> 5, lane = threadIdx.x & 31;
  const int half = lane >> 4, l16 = lane & 15;

  // ---- stage 1: scores via WMMA (16 tiles over 8 waves) ----
  for (int tile = wave; tile < 16; tile += 8) {
    const int tmq = tile >> 2;              // query row tile 0..3
    const int tnk = tile & 3;               // key  col tile 0..3
    int arow = tmq * 16 + l16; if (arow > kD - 1) arow = kD - 1;  // clamp pad
    int bcol = tnk * 16 + l16; if (bcol > kD - 1) bcol = kD - 1;
    const bf16* arp = qs + (size_t)arow * kH;
    const bf16* brp = ks + (size_t)bcol * kH;   // K^T column jj == k2 row jj
    // two accumulators (even/odd k-steps) + ping-pong prefetch, fully unrolled
    v8f acc0 = {}, acc1 = {};
    v16bf a0 = load_afrag(arp, 0, half), b0 = load_bfrag(brp, 0, half);
    v16bf a1, b1;
#pragma unroll
    for (int kb = 0; kb < kH - 64; kb += 64) {
      a1 = load_afrag(arp, kb + 32, half); b1 = load_bfrag(brp, kb + 32, half);
      acc0 = wmma_bf16(a0, b0, acc0);
      a0 = load_afrag(arp, kb + 64, half); b0 = load_bfrag(brp, kb + 64, half);
      acc1 = wmma_bf16(a1, b1, acc1);
    }
    a1 = load_afrag(arp, kH - 32, half); b1 = load_bfrag(brp, kH - 32, half);
    acc0 = wmma_bf16(a0, b0, acc0);
    acc1 = wmma_bf16(a1, b1, acc1);
#pragma unroll
    for (int r = 0; r < 8; ++r)
      S[(tmq * 16 + half * 8 + r) * kSS + tnk * 16 + l16] =
          (acc0[r] + acc1[r]) * kScale;
  }
  __syncthreads();

  // ---- stage 2: softmax rows 0..48; probs -> bf16, padded cols/rows zero ----
  if (threadIdx.x < 64) {
    const int t = threadIdx.x;
    bf16* prow = Pb + t * kPS;
    if (t < kD) {
      float* row = S + t * kSS;
      float mx = -1e30f;
      for (int c = 0; c < kD; ++c) mx = fmaxf(mx, row[c]);
      float sm = 0.f;
      for (int c = 0; c < kD; ++c) { float e = expf(row[c] - mx); row[c] = e; sm += e; }
      const float inv = 1.f / sm;
      for (int c = 0; c < kD; ++c) prow[c] = (bf16)(row[c] * inv);
      for (int c = kD; c < kKP; ++c) prow[c] = (bf16)0.f;
    } else {
      for (int c = 0; c < kKP; ++c) prow[c] = (bf16)0.f;
    }
  }
  __syncthreads();

  // ---- stage 3: Out = P @ V via WMMA (4 x 32 tiles over 8 waves) ----
  for (int tile = wave; tile < 128; tile += 8) {
    const int tmq = tile >> 5;              // 0..3
    const int tne = tile & 31;              // 0..31 channel tile
    const bf16* arp = Pb + (size_t)(tmq * 16 + l16) * kPS;      // LDS
    const bf16* brp = vt + (size_t)(tne * 16 + l16) * kKP;      // global
    // K = 64: load both fragment pairs up front, then two WMMAs
    v16bf a0 = load_afrag(arp, 0, half),  b0 = load_bfrag(brp, 0, half);
    v16bf a1 = load_afrag(arp, 32, half), b1 = load_bfrag(brp, 32, half);
    v8f acc = {};
    acc = wmma_bf16(a0, b0, acc);
    acc = wmma_bf16(a1, b1, acc);
    const int ecol = tne * 16 + l16;
#pragma unroll
    for (int r = 0; r < 8; ++r) {
      const int row = tmq * 16 + half * 8 + r;
      if (row < kD) os[(size_t)row * kC + ecol] = (bf16)acc[r];
    }
  }
}

// ---------------------------------------------------------------------------
// BatchNorm over channel axis (144 channels, stats over B*D*C elems each).
// ---------------------------------------------------------------------------
__global__ __launch_bounds__(256)
void bn_partial(const float* __restrict__ edge, float* __restrict__ partials) {
  const int SL = kD * kC;
  const float* base = edge + (size_t)blockIdx.x * SL;
  float s = 0.f, s2 = 0.f;
  for (int idx = threadIdx.x; idx < SL; idx += 256) {
    const float v = base[idx];
    s += v; s2 += v * v;
  }
  __shared__ float sh[256], sh2[256];
  sh[threadIdx.x] = s; sh2[threadIdx.x] = s2;
  __syncthreads();
  for (int off = 128; off > 0; off >>= 1) {
    if (threadIdx.x < off) {
      sh[threadIdx.x]  += sh[threadIdx.x + off];
      sh2[threadIdx.x] += sh2[threadIdx.x + off];
    }
    __syncthreads();
  }
  if (threadIdx.x == 0) {
    partials[2 * blockIdx.x]     = sh[0];
    partials[2 * blockIdx.x + 1] = sh2[0];
  }
}

__global__ __launch_bounds__(256)
void bn_final(const float* __restrict__ partials, const float* __restrict__ bn_w,
              const float* __restrict__ bn_b, float* __restrict__ chp) {
  const int p = threadIdx.x;
  if (p >= kP) return;
  float s = 0.f, s2 = 0.f;
  for (int b = 0; b < kB; ++b) {
    const int idx = b * kP + p;
    s += partials[2 * idx]; s2 += partials[2 * idx + 1];
  }
  const float invn = 1.f / ((float)kB * kD * kC);
  const float mean = s * invn;
  const float var = s2 * invn - mean * mean;
  const float scale = bn_w[p] * rsqrtf(var + kEps);
  chp[2 * p] = scale;
  chp[2 * p + 1] = bn_b[p] - mean * scale;
}

__global__ __launch_bounds__(256)
void bn_apply(float* __restrict__ edge, const float* __restrict__ chp) {
  const int p = blockIdx.x % kP;
  const float scale = chp[2 * p], shift = chp[2 * p + 1];
  float* base = edge + (size_t)blockIdx.x * (kD * kC);
  for (int idx = threadIdx.x; idx < kD * kC; idx += 256)
    base[idx] = base[idx] * scale + shift;
}

// ---------------------------------------------------------------------------
// Host launcher
// ---------------------------------------------------------------------------
extern "C" void kernel_launch(void* const* d_in, const int* in_sizes, int n_in,
                              void* d_out, int out_size, void* d_ws, size_t ws_size,
                              hipStream_t stream) {
  const float* cf  = (const float*)d_in[0];   // (B,N,D,C)
  const float* gf  = (const float*)d_in[1];   // (B,D,C)
  const float* fqw = (const float*)d_in[2];
  const float* fqb = (const float*)d_in[3];
  const float* fkw = (const float*)d_in[4];
  const float* fkb = (const float*)d_in[5];
  const float* fvw = (const float*)d_in[6];
  const float* fvb = (const float*)d_in[7];
  const float* aqw = (const float*)d_in[8];
  const float* aqb = (const float*)d_in[9];
  const float* akw = (const float*)d_in[10];
  const float* akb = (const float*)d_in[11];
  const float* avw = (const float*)d_in[12];
  const float* avb = (const float*)d_in[13];
  const float* ew  = (const float*)d_in[14];
  const float* eb  = (const float*)d_in[15];
  const float* bnw = (const float*)d_in[16];
  const float* bnb = (const float*)d_in[17];
  float* out = (float*)d_out;                 // (B, N*N, D, C)

  // workspace carve-up (~200 MB)
  char* ws = (char*)d_ws;
  size_t off = 0;
  auto alloc = [&](size_t bytes) -> void* {
    off = (off + 255) & ~(size_t)255;
    void* p = ws + off;
    off += bytes;
    return p;
  };
  const size_t nCF = (size_t)kB * kN * kD * kC;   // 4,816,896
  const size_t nGF = (size_t)kB * kD * kC;        //   401,408
  const size_t nQ2 = (size_t)kB * kN * kD * kH;   // 2,408,448
  const size_t nF2 = (size_t)kB * kP * kD * kC;   // 57,802,752

  bf16* cf_bf   = (bf16*)alloc(nCF * 2);
  bf16* gf_bf   = (bf16*)alloc(nGF * 2);
  bf16* fqw_bf  = (bf16*)alloc((size_t)kH * kC * 2);
  bf16* fkw_bf  = (bf16*)alloc((size_t)kH * kC * 2);
  bf16* fvw_bf  = (bf16*)alloc((size_t)kC * kC * 2);
  bf16* aqw_bf  = (bf16*)alloc((size_t)kH * kC * 2);
  bf16* akw_bf  = (bf16*)alloc((size_t)kH * kC * 2);
  bf16* avw_bf  = (bf16*)alloc((size_t)kC * kC * 2);
  bf16* ew_bf   = (bf16*)alloc((size_t)kC * kC * 2);
  float* q1     = (float*)alloc(nQ2 * 4);
  float* k1     = (float*)alloc((size_t)kB * kD * kH * 4);
  float* v1     = (float*)alloc(nGF * 4);
  bf16* feat_bf = (bf16*)alloc(nCF * 2);
  bf16* q2b     = (bf16*)alloc(nQ2 * 2);
  bf16* k2b     = (bf16*)alloc(nQ2 * 2);
  float* v2     = (float*)alloc(nCF * 4);
  bf16* v2t     = (bf16*)alloc((size_t)kB * kN * kC * kKP * 2);
  bf16* f2_bf   = (bf16*)alloc(nF2 * 2);
  float* partials = (float*)alloc((size_t)kB * kP * 2 * 4);
  float* chp      = (float*)alloc((size_t)kP * 2 * 4);
  (void)ws_size; (void)n_in; (void)in_sizes; (void)out_size;

  auto cvt = [&](const float* s, bf16* d, size_t n) {
    cvt_bf16<<<dim3((unsigned)((n + 255) / 256)), dim3(256), 0, stream>>>(s, d, (int)n);
  };
  auto gemm = [&](const bf16* X, const bf16* W, const float* bias, float* Y,
                  bf16* Ybf, int M, int Nn) {
    dim3 grid((unsigned)(Nn / 64), (unsigned)(((M / 16) + 7) / 8));
    gemm_bf16<<<grid, dim3(256), 0, stream>>>(X, W, bias, Y, Ybf, M, Nn);
  };

  // 1) convert activations + weights to bf16
  cvt(cf, cf_bf, nCF);
  cvt(gf, gf_bf, nGF);
  cvt(fqw, fqw_bf, (size_t)kH * kC);
  cvt(fkw, fkw_bf, (size_t)kH * kC);
  cvt(fvw, fvw_bf, (size_t)kC * kC);
  cvt(aqw, aqw_bf, (size_t)kH * kC);
  cvt(akw, akw_bf, (size_t)kH * kC);
  cvt(avw, avw_bf, (size_t)kC * kC);
  cvt(ew,  ew_bf,  (size_t)kC * kC);

  // 2) FAM projections (WMMA)
  gemm(cf_bf, fqw_bf, fqb, q1, nullptr, kB * kN * kD, kH);   // 9408 x 256
  gemm(gf_bf, fkw_bf, fkb, k1, nullptr, kB * kD, kH);        //  784 x 256
  gemm(gf_bf, fvw_bf, fvb, v1, nullptr, kB * kD, kC);        //  784 x 512

  // 3) FAM attention -> feat (bf16)
  fam_attn<<<dim3(kB * kN * kD), dim3(256), 0, stream>>>(q1, k1, v1, feat_bf);

  // 4) ARM projections (WMMA); q2/k2 straight to bf16, v2 to f32 for transpose
  gemm(feat_bf, aqw_bf, aqb, nullptr, q2b, kB * kN * kD, kH);
  gemm(feat_bf, akw_bf, akb, nullptr, k2b, kB * kN * kD, kH);
  gemm(feat_bf, avw_bf, avb, v2, nullptr, kB * kN * kD, kC);
  transpose_v2<<<dim3(kB * kN), dim3(256), 0, stream>>>(v2, v2t);

  // 5) ARM pairwise attention (WMMA) -> feat2 (bf16)
  arm_attn_wmma<<<dim3(kB * kP), dim3(256), 0, stream>>>(q2b, k2b, v2t, f2_bf);

  // 6) edge GEMM (WMMA) straight into d_out (pre-BN)
  gemm(f2_bf, ew_bf, eb, out, nullptr, kB * kP * kD, kC);    // 112896 x 512

  // 7) BatchNorm (deterministic two-level reduction, in-place apply)
  bn_partial<<<dim3(kB * kP), dim3(256), 0, stream>>>(out, partials);
  bn_final<<<dim3(1), dim3(256), 0, stream>>>(partials, bnw, bnb, chp);
  bn_apply<<<dim3(kB * kP), dim3(256), 0, stream>>>(out, chp);
}
